// HRM_23794118820172
// MI455X (gfx1250) — compile-verified, matrix operand order checked by reading, not run on hardware
//
#include <hip/hip_runtime.h>
#include <hip/hip_bf16.h>
#include <math.h>

typedef __attribute__((ext_vector_type(16))) _Float16 v16h;
typedef __attribute__((ext_vector_type(8)))  _Float16 h8;
typedef __attribute__((ext_vector_type(8)))  float    v8f;
typedef __attribute__((ext_vector_type(4)))  float    f4;

#define ACT_NONE 0
#define ACT_GELU 1
#define ACT_SIG  2
#define ACT_TANH 3

// B operand is pre-converted f16, TRANSPOSED [N][K_full]; W points at the
// sub-block (base + row-offset in the original matrix), ldw = K_full.
struct Pair { const float* A; const _Float16* W; int K; int lda; int ldw; };

__device__ __forceinline__ float act_apply(float x, int act) {
  if (act == ACT_GELU) return 0.5f * x * (1.0f + erff(x * 0.7071067811865476f));
  if (act == ACT_SIG)  return 1.0f / (1.0f + expf(-x));
  if (act == ACT_TANH) return tanhf(x);
  return x;
}

// Accumulate one (A,W) pair. Rows/cols are pre-clamped: out-of-range tile
// rows/cols load real (in-bounds) data and their results are masked at the
// store, so the inner loop has NO predication. Requires K % 32 == 0.
// Fragment layout (wave32, 16-bit A 16x32 / B 32x16, ISA 7.12.2):
// lanes 0-15: elems 0..7 -> K k0+0..7,  elems 8..15 -> K k0+16..23
// lanes16-31: elems 0..7 -> K k0+8..15, elems 8..15 -> K k0+24..31
__device__ __forceinline__ v8f gemm_accum(Pair P, v8f acc, int m_c, int n_c,
                                          int half) {
  const float*    arow = P.A + (long)m_c * P.lda;
  const _Float16* brow = P.W + (long)n_c * P.ldw;
  for (int k0 = 0; k0 < P.K; k0 += 32) {
    const int ka = k0 + 8 * half;
    // A: two contiguous 8-float runs -> 4x b128 loads, packed cvt to f16
    f4 a0 = *(const f4*)(arow + ka);
    f4 a1 = *(const f4*)(arow + ka + 4);
    f4 a2 = *(const f4*)(arow + ka + 16);
    f4 a3 = *(const f4*)(arow + ka + 20);
    // B: two contiguous 8-half runs -> 2x b128 loads (transposed f16 weights)
    h8 blo = *(const h8*)(brow + ka);
    h8 bhi = *(const h8*)(brow + ka + 16);
    v16h a, b;
#pragma unroll
    for (int e = 0; e < 4; ++e) {
      a[e]      = (_Float16)a0[e];
      a[4 + e]  = (_Float16)a1[e];
      a[8 + e]  = (_Float16)a2[e];
      a[12 + e] = (_Float16)a3[e];
    }
#pragma unroll
    for (int e = 0; e < 8; ++e) { b[e] = blo[e]; b[8 + e] = bhi[e]; }
    acc = __builtin_amdgcn_wmma_f32_16x16x32_f16(false, a, false, b,
                                                 (short)0, acc, false, false);
  }
  return acc;
}

// Generic multi-operand GEMM: C = act( sum_i A_i @ W_i + bias0 + bias1 + Madd )
// One wave per 16x16 output tile; concatenated inputs become multiple (A,W)
// pairs sharing one accumulator; lda==0 broadcasts a row.
__global__ __launch_bounds__(32)
void k_gemm(Pair p0, Pair p1, Pair p2, int npairs,
            const float* __restrict__ bias0, const float* __restrict__ bias1,
            const float* __restrict__ Madd, int ldm,
            float* __restrict__ C, int ldc, int M, int N, int act)
{
  const int lane = threadIdx.x;
  const int half = lane >> 4;
  const int mn   = lane & 15;
  const int tm = blockIdx.x * 16;
  const int tn = blockIdx.y * 16;
  const int m = tm + mn;
  const int n = tn + mn;
  const int m_c = (m < M) ? m : (M - 1);   // clamp: garbage masked at store
  const int n_c = (n < N) ? n : (N - 1);
  v8f acc = {};
  acc = gemm_accum(p0, acc, m_c, n_c, half);
  if (npairs > 1) acc = gemm_accum(p1, acc, m_c, n_c, half);
  if (npairs > 2) acc = gemm_accum(p2, acc, m_c, n_c, half);
#pragma unroll
  for (int r = 0; r < 8; ++r) {
    int row = tm + r + 8 * half;   // C/D layout: VGPR r -> M = r + 8*(lane>=16)
    int col = tn + mn;
    if (row < M && col < N) {
      float v = acc[r];
      if (bias0) v += bias0[col];
      if (bias1) v += bias1[col];
      if (Madd)  v += Madd[(long)row * ldm + col];
      C[(long)row * ldc + col] = act_apply(v, act);
    }
  }
}

// fp32 W (K x N, row-major) -> f16 transposed Wt (N x K, row-major)
__global__ void k_cvtWt(const float* __restrict__ W, _Float16* __restrict__ Wt,
                        int K, int N) {
  int i = blockIdx.x * 256 + threadIdx.x;
  if (i < K * N) {
    int n = i / K, k = i - n * K;
    Wt[i] = (_Float16)W[(long)k * N + n];
  }
}

// diff[b,k,:] = mean_L(tok[sup_out]-tok[sup_inp]) via signed color histogram.
__global__ __launch_bounds__(256)
void k_diff(const int* __restrict__ sup_in, const int* __restrict__ sup_out,
            const float* __restrict__ tok, float* __restrict__ diff, int Llen)
{
  __shared__ int hist[10];
  const int bk = blockIdx.x, tid = threadIdx.x;
  if (tid < 10) hist[tid] = 0;
  __syncthreads();
  const int* pin  = sup_in  + (long)bk * Llen;
  const int* pout = sup_out + (long)bk * Llen;
  for (int l = tid; l < Llen; l += 256) {
    atomicAdd(&hist[pout[l]], 1);
    atomicSub(&hist[pin[l]], 1);
  }
  __syncthreads();
  if (tid < 128) {
    float acc = 0.0f;
    for (int c = 0; c < 10; ++c) acc += (float)hist[c] * tok[c * 128 + tid];
    diff[(long)bk * 128 + tid] = acc / (float)Llen;
  }
}

// q_pool[b,:] = histogram(q[b])@tok / L + mean(pos)
__global__ __launch_bounds__(256)
void k_qpool(const int* __restrict__ q, const float* __restrict__ tok,
             const float* __restrict__ posmean, float* __restrict__ qpool, int Llen)
{
  __shared__ int hist[10];
  const int b = blockIdx.x, tid = threadIdx.x;
  if (tid < 10) hist[tid] = 0;
  __syncthreads();
  const int* p = q + (long)b * Llen;
  for (int l = tid; l < Llen; l += 256) atomicAdd(&hist[p[l]], 1);
  __syncthreads();
  if (tid < 128) {
    float acc = 0.0f;
    for (int c = 0; c < 10; ++c) acc += (float)hist[c] * tok[c * 128 + tid];
    qpool[(long)b * 128 + tid] = acc / (float)Llen + posmean[tid];
  }
}

__global__ __launch_bounds__(128)
void k_posmean(const float* __restrict__ pos, float* __restrict__ pm, int Llen) {
  const int d = threadIdx.x;
  float acc = 0.0f;
  for (int l = 0; l < Llen; ++l) acc += pos[(long)l * 128 + d];
  pm[d] = acc / (float)Llen;
}

__global__ void k_mean3(const float* __restrict__ x, float* __restrict__ y, int n) {
  int i = blockIdx.x * 256 + threadIdx.x;
  if (i < n) {
    int b = i >> 7, d = i & 127;
    y[i] = (x[(long)(3 * b) * 128 + d] + x[(long)(3 * b + 1) * 128 + d] +
            x[(long)(3 * b + 2) * 128 + d]) * (1.0f / 3.0f);
  }
}

// state = (1-g)*state + g*c  (cell update; in-place elementwise)
__global__ void k_cellupd(float* __restrict__ s, const float* __restrict__ g,
                          const float* __restrict__ c, int n) {
  int i = blockIdx.x * 256 + threadIdx.x;
  if (i < n) { float gv = g[i]; s[i] = (1.0f - gv) * s[i] + gv * c[i]; }
}

// out_head W2 (128x10 fp32) -> TRANSPOSED (16x128 f16, zero-padded rows) so
// decode B-fragments are contiguous 8-half runs.
__global__ __launch_bounds__(256)
void k_cvtw2(const float* __restrict__ W2, _Float16* __restrict__ W2hT) {
  for (int i = threadIdx.x; i < 16 * 128; i += 256) {
    int c = i >> 7, k = i & 127;
    W2hT[i] = (c < 10) ? (_Float16)W2[k * 10 + c] : (_Float16)0.0f;
  }
}

// Fused decode: h1 = tokW1[q] + posW1[l] + rowbias[b]; gelu -> LDS f16;
// logits = gelu(h1) @ W2^T  (K=128 as 4 chained WMMAs, N padded 10->16).
// 8 independent waves per block, one 16-position tile + 4KB LDS slice each.
__global__ __launch_bounds__(256)
void k_decode(const int* __restrict__ qidx, const float* __restrict__ tokW1,
              const float* __restrict__ posW1, const float* __restrict__ rowb,
              const _Float16* __restrict__ W2hT, const float* __restrict__ b2,
              float* __restrict__ out, int Llen)
{
  __shared__ _Float16 hbuf[8][16 * 128];
  const int wave = threadIdx.x >> 5;
  const int lane = threadIdx.x & 31;
  const int half = lane >> 4;
  const int mn   = lane & 15;
  _Float16* hb = &hbuf[wave][0];
  const long base = ((long)blockIdx.x * 8 + wave) * 16;  // flat into (B*L)

  // ---- phase 1: 2 lanes per position, each covers 64 of 128 dims ----
  {
    const long flat = base + mn;
    const int  b = (int)(flat / Llen);
    const int  l = (int)(flat % Llen);
    const int  c = qidx[flat];
    const float* tR = tokW1 + c * 128;
    const float* pR = posW1 + (long)l * 128;
    const float* rR = rowb  + (long)b * 128;
    _Float16* hrow = hb + mn * 128;
#pragma unroll
    for (int it = 0; it < 4; ++it) {
      const int d0 = half * 64 + it * 16;
      h8 outlo, outhi;
#pragma unroll
      for (int g = 0; g < 4; ++g) {
        f4 t = *(const f4*)(tR + d0 + 4 * g);
        f4 p = *(const f4*)(pR + d0 + 4 * g);
        f4 r = *(const f4*)(rR + d0 + 4 * g);
#pragma unroll
        for (int e = 0; e < 4; ++e) {
          float v = t[e] + p[e] + r[e];
          v = 0.5f * v * (1.0f + erff(v * 0.7071067811865476f));
          if (g < 2) outlo[4 * g + e] = (_Float16)v;
          else       outhi[4 * (g - 2) + e] = (_Float16)v;
        }
      }
      *(h8*)(hrow + d0)     = outlo;
      *(h8*)(hrow + d0 + 8) = outhi;
    }
  }
  __syncthreads();

  // ---- phase 2: 4x v_wmma_f32_16x16x32_f16 over K=128 ----
  v8f acc = {};
  for (int k0 = 0; k0 < 128; k0 += 32) {
    const int ka = k0 + 8 * half;
    h8 alo = *(const h8*)(hb + mn * 128 + ka);        // ds_load_b128
    h8 ahi = *(const h8*)(hb + mn * 128 + ka + 16);
    h8 blo = *(const h8*)(W2hT + mn * 128 + ka);      // global b128 (L0-hot)
    h8 bhi = *(const h8*)(W2hT + mn * 128 + ka + 16);
    v16h a, b;
#pragma unroll
    for (int e = 0; e < 8; ++e) {
      a[e] = alo[e]; a[8 + e] = ahi[e];
      b[e] = blo[e]; b[8 + e] = bhi[e];
    }
    acc = __builtin_amdgcn_wmma_f32_16x16x32_f16(false, a, false, b,
                                                 (short)0, acc, false, false);
  }
#pragma unroll
  for (int r = 0; r < 8; ++r) {
    int p = r + 8 * half;
    if (mn < 10) out[(base + p) * 10 + mn] = acc[r] + b2[mn];
  }
}

static inline int cdiv16(int x) { return (x + 15) >> 4; }

static void gemm(hipStream_t s, int M, int N,
                 Pair p0, Pair p1, Pair p2, int np,
                 const float* b0, const float* b1,
                 const float* Madd, int ldm,
                 float* C, int ldc, int act) {
  dim3 g((unsigned)cdiv16(M), (unsigned)cdiv16(N));
  k_gemm<<<g, 32, 0, s>>>(p0, p1, p2, np, b0, b1, Madd, ldm, C, ldc, M, N, act);
}

extern "C" void kernel_launch(void* const* d_in, const int* in_sizes, int n_in,
                              void* d_out, int out_size, void* d_ws, size_t ws_size,
                              hipStream_t stream)
{
  (void)in_sizes; (void)n_in; (void)out_size; (void)ws_size;
  const int B = 256, L = 900;
  const int DM = 128, DL = 160, DH = 192, NC = 10, CYC = 4, STEPS = 3;

  const int* sup_in  = (const int*)d_in[0];
  const int* sup_out = (const int*)d_in[1];
  const int* query   = (const int*)d_in[2];
  // params flattened as JAX pytree (sorted dict keys at every level)
  const float* agg_W1 = (const float*)d_in[3];
  const float* agg_W2 = (const float*)d_in[4];
  const float* agg_b1 = (const float*)d_in[5];
  const float* agg_b2 = (const float*)d_in[6];
  const float* cp_W1  = (const float*)d_in[7];
  const float* cp_W2  = (const float*)d_in[8];
  const float* cp_b1  = (const float*)d_in[9];
  const float* cp_b2  = (const float*)d_in[10];
  const float* hc_Wic = (const float*)d_in[11];
  const float* hc_Wig = (const float*)d_in[12];
  const float* hc_Wsc = (const float*)d_in[13];
  const float* hc_Wsg = (const float*)d_in[14];
  const float* hc_bic = (const float*)d_in[15];
  const float* hc_big = (const float*)d_in[16];
  const float* hc_bsc = (const float*)d_in[17];
  const float* hc_bsg = (const float*)d_in[18];
  const float* hi_W1  = (const float*)d_in[19];
  const float* hi_W2  = (const float*)d_in[20];
  const float* hi_b1  = (const float*)d_in[21];
  const float* hi_b2  = (const float*)d_in[22];
  const float* ha_W1  = (const float*)d_in[23];
  const float* ha_W2  = (const float*)d_in[24];
  const float* ha_b1  = (const float*)d_in[25];
  const float* ha_b2  = (const float*)d_in[26];
  const float* lc_Wic = (const float*)d_in[27];
  const float* lc_Wig = (const float*)d_in[28];
  const float* lc_Wsc = (const float*)d_in[29];
  const float* lc_Wsg = (const float*)d_in[30];
  const float* lc_bic = (const float*)d_in[31];
  const float* lc_big = (const float*)d_in[32];
  const float* lc_bsc = (const float*)d_in[33];
  const float* lc_bsg = (const float*)d_in[34];
  const float* li_W1  = (const float*)d_in[35];
  const float* li_W2  = (const float*)d_in[36];
  const float* li_b1  = (const float*)d_in[37];
  const float* li_b2  = (const float*)d_in[38];
  const float* oh_W1  = (const float*)d_in[39];
  const float* oh_W2  = (const float*)d_in[40];
  const float* oh_b1  = (const float*)d_in[41];
  const float* oh_b2  = (const float*)d_in[42];
  const float* pos    = (const float*)d_in[43];
  const float* task   = (const float*)d_in[44];
  const float* tok    = (const float*)d_in[45];

  float* ws = (float*)d_ws;
  size_t o = 0;
  float* posmean = ws + o; o += 128;
  float* diff    = ws + o; o += (size_t)B * 3 * DM;
  float* qpool   = ws + o; o += (size_t)B * DM;
  float* bufA    = ws + o; o += (size_t)B * 3 * DM;
  float* pairout = ws + o; o += (size_t)B * 3 * DM;
  float* sctxin  = ws + o; o += (size_t)B * DM;
  float* sctx    = ws + o; o += (size_t)B * DM;
  float* htmp    = ws + o; o += (size_t)B * DH;
  float* H       = ws + o; o += (size_t)B * DH;
  float* Lb      = ws + o; o += (size_t)B * DL;
  float* Gg      = ws + o; o += (size_t)B * DL;
  float* Gc      = ws + o; o += (size_t)B * DL;
  float* gb      = ws + o; o += (size_t)B * DH;
  float* cb      = ws + o; o += (size_t)B * DH;
  float* rowb    = ws + o; o += (size_t)B * DM;
  float* tokW1   = ws + o; o += (size_t)16 * DM;
  float* posW1   = ws + o; o += (size_t)L * DM;

  // f16 arena (all sizes multiples of 8 halves -> 16B alignment preserved)
  _Float16* hws = (_Float16*)(ws + o);
  size_t ho = 0;
  _Float16* W2hT = hws + ho; ho += 16 * 128;
  struct CvtJob { const float* W; _Float16* Wt; int K, N; };
  _Float16 *cpW1t, *cpW2t, *aggW1t, *aggW2t, *hiW1t, *hiW2t, *liW1t, *liW2t;
  _Float16 *lcWigT, *lcWicT, *lcWsgT, *lcWscT, *hcWigT, *hcWicT, *hcWsgT, *hcWscT;
  _Float16 *ohW1t, *haW1t, *haW2t;
  cpW1t  = hws + ho; ho += 256 * DM;
  cpW2t  = hws + ho; ho += DM * DM;
  aggW1t = hws + ho; ho += DM * DM;
  aggW2t = hws + ho; ho += DM * DM;
  hiW1t  = hws + ho; ho += 256 * DH;
  hiW2t  = hws + ho; ho += DH * DH;
  liW1t  = hws + ho; ho += 448 * DL;
  liW2t  = hws + ho; ho += DL * DL;
  lcWigT = hws + ho; ho += 480 * DL;
  lcWicT = hws + ho; ho += 480 * DL;
  lcWsgT = hws + ho; ho += DL * DL;
  lcWscT = hws + ho; ho += DL * DL;
  hcWigT = hws + ho; ho += 352 * DH;
  hcWicT = hws + ho; ho += 352 * DH;
  hcWsgT = hws + ho; ho += DH * DH;
  hcWscT = hws + ho; ho += DH * DH;
  ohW1t  = hws + ho; ho += 448 * DM;
  haW1t  = hws + ho; ho += DH * 96;
  haW2t  = hws + ho; ho += 96 * 2;

  // --- one-time weight conversion to transposed f16 ---
  const CvtJob jobs[] = {
    {cp_W1, cpW1t, 256, DM}, {cp_W2, cpW2t, DM, DM},
    {agg_W1, aggW1t, DM, DM}, {agg_W2, aggW2t, DM, DM},
    {hi_W1, hiW1t, 256, DH}, {hi_W2, hiW2t, DH, DH},
    {li_W1, liW1t, 448, DL}, {li_W2, liW2t, DL, DL},
    {lc_Wig, lcWigT, 480, DL}, {lc_Wic, lcWicT, 480, DL},
    {lc_Wsg, lcWsgT, DL, DL}, {lc_Wsc, lcWscT, DL, DL},
    {hc_Wig, hcWigT, 352, DH}, {hc_Wic, hcWicT, 352, DH},
    {hc_Wsg, hcWsgT, DH, DH}, {hc_Wsc, hcWscT, DH, DH},
    {oh_W1, ohW1t, 448, DM}, {ha_W1, haW1t, DH, 96}, {ha_W2, haW2t, 96, 2},
  };
  for (const CvtJob& j : jobs) {
    int n = j.K * j.N;
    k_cvtWt<<<(n + 255) / 256, 256, 0, stream>>>(j.W, j.Wt, j.K, j.N);
  }
  k_cvtw2<<<1, 256, 0, stream>>>(oh_W2, W2hT);

  // --- support/query pooling (positional embedding cancels in diff) ---
  k_posmean<<<1, 128, 0, stream>>>(pos, posmean, L);
  k_diff<<<B * 3, 256, 0, stream>>>(sup_in, sup_out, tok, diff, L);
  k_qpool<<<B, 256, 0, stream>>>(query, tok, posmean, qpool, L);

  // --- ctx_pair MLP (concat[diff,diff] => two pairs on same A) ---
  gemm(stream, B * 3, DM, Pair{diff, cpW1t, DM, DM, 256},
       Pair{diff, cpW1t + DM, DM, DM, 256}, Pair{}, 2,
       cp_b1, nullptr, nullptr, 0, bufA, DM, ACT_GELU);
  gemm(stream, B * 3, DM, Pair{bufA, cpW2t, DM, DM, DM}, Pair{}, Pair{}, 1,
       cp_b2, nullptr, nullptr, 0, pairout, DM, ACT_NONE);
  { int n = B * DM; k_mean3<<<(n + 255) / 256, 256, 0, stream>>>(pairout, sctxin, n); }
  gemm(stream, B, DM, Pair{sctxin, aggW1t, DM, DM, DM}, Pair{}, Pair{}, 1,
       agg_b1, nullptr, nullptr, 0, htmp, DM, ACT_GELU);
  gemm(stream, B, DM, Pair{htmp, aggW2t, DM, DM, DM}, Pair{}, Pair{}, 1,
       agg_b2, nullptr, nullptr, 0, sctx, DM, ACT_NONE);

  // --- h_init (task token row-broadcast via lda=0) ---
  gemm(stream, B, DH, Pair{task, hiW1t, DM, 0, 256},
       Pair{sctx, hiW1t + DM, DM, DM, 256}, Pair{}, 2,
       hi_b1, nullptr, nullptr, 0, htmp, DH, ACT_GELU);
  gemm(stream, B, DH, Pair{htmp, hiW2t, DH, DH, DH}, Pair{}, Pair{}, 1,
       hi_b2, nullptr, nullptr, 0, H, DH, ACT_NONE);

  // --- decode cycle-invariant tables: tok@W1a, pos@W1a ---
  gemm(stream, NC, DM, Pair{tok, ohW1t, DM, DM, 448}, Pair{}, Pair{}, 1,
       nullptr, nullptr, nullptr, 0, tokW1, DM, ACT_NONE);
  gemm(stream, L, DM, Pair{pos, ohW1t, DM, DM, 448}, Pair{}, Pair{}, 1,
       nullptr, nullptr, nullptr, 0, posW1, DM, ACT_NONE);

  float* out = (float*)d_out;
  const size_t LOGITS = (size_t)CYC * B * L * NC;

  for (int cyc = 0; cyc < CYC; ++cyc) {
    // l_init on concat(H, sup_ctx, q_pool)
    gemm(stream, B, DL, Pair{H, liW1t, DH, DH, 448},
         Pair{sctx, liW1t + DH, DM, DM, 448},
         Pair{qpool, liW1t + DH + DM, DM, DM, 448}, 3,
         li_b1, nullptr, nullptr, 0, htmp, DL, ACT_GELU);
    gemm(stream, B, DL, Pair{htmp, liW2t, DL, DL, DL}, Pair{}, Pair{}, 1,
         li_b2, nullptr, nullptr, 0, Lb, DL, ACT_NONE);

    // inner-step-invariant gate/cand terms: H@Wig_H + qpool@Wig_q + big + bsg
    gemm(stream, B, DL, Pair{H, lcWigT + DL, DH, DH, 480},
         Pair{qpool, lcWigT + DL + DH, DM, DM, 480}, Pair{}, 2,
         lc_big, lc_bsg, nullptr, 0, Gg, DL, ACT_NONE);
    gemm(stream, B, DL, Pair{H, lcWicT + DL, DH, DH, 480},
         Pair{qpool, lcWicT + DL + DH, DM, DM, 480}, Pair{}, 2,
         lc_bic, lc_bsc, nullptr, 0, Gc, DL, ACT_NONE);

    for (int t = 0; t < STEPS; ++t) {
      gemm(stream, B, DL, Pair{Lb, lcWigT, DL, DL, 480},
           Pair{Lb, lcWsgT, DL, DL, DL}, Pair{}, 2,
           nullptr, nullptr, Gg, DL, gb, DL, ACT_SIG);
      gemm(stream, B, DL, Pair{Lb, lcWicT, DL, DL, 480},
           Pair{Lb, lcWscT, DL, DL, DL}, Pair{}, 2,
           nullptr, nullptr, Gc, DL, cb, DL, ACT_TANH);
      int n = B * DL;
      k_cellupd<<<(n + 255) / 256, 256, 0, stream>>>(Lb, gb, cb, n);
    }

    // h_cell on concat(H, L) with prev=H recurrent term
    gemm(stream, B, DH, Pair{H, hcWigT, DH, DH, 352},
         Pair{Lb, hcWigT + DH, DL, DL, 352},
         Pair{H, hcWsgT, DH, DH, DH}, 3,
         hc_big, hc_bsg, nullptr, 0, gb, DH, ACT_SIG);
    gemm(stream, B, DH, Pair{H, hcWicT, DH, DH, 352},
         Pair{Lb, hcWicT + DH, DL, DL, 352},
         Pair{H, hcWscT, DH, DH, DH}, 3,
         hc_bic, hc_bsc, nullptr, 0, cb, DH, ACT_TANH);
    { int n = B * DH; k_cellupd<<<(n + 255) / 256, 256, 0, stream>>>(H, gb, cb, n); }

    // per-(cycle,b) decode bias: H@W1b + sup_ctx@W1c + b1
    gemm(stream, B, DM, Pair{H, ohW1t + DM, DH, DH, 448},
         Pair{sctx, ohW1t + DM + DH, DM, DM, 448}, Pair{}, 2,
         oh_b1, nullptr, nullptr, 0, rowb, DM, ACT_NONE);

    // fused decode: 1800 blocks x 8 waves, one 16-position tile per wave
    k_decode<<<(B * L) / (16 * 8), 256, 0, stream>>>(query, tokW1, posW1, rowb,
                                                     W2hT, oh_b2,
                                                     out + (size_t)cyc * B * L * NC, L);

    // halt head -> directly into d_out q region
    gemm(stream, B, 96, Pair{H, haW1t, DH, DH, DH}, Pair{}, Pair{}, 1,
         ha_b1, nullptr, nullptr, 0, htmp, 96, ACT_GELU);
    gemm(stream, B, 2, Pair{htmp, haW2t, 96, 96, 96}, Pair{}, Pair{}, 1,
         ha_b2, nullptr, nullptr, 0, out + LOGITS + (size_t)cyc * B * 2, 2, ACT_NONE);
  }

  // final H
  hipMemcpyAsync(out + LOGITS + (size_t)CYC * B * 2, H,
                 (size_t)B * DH * sizeof(float), hipMemcpyDeviceToDevice, stream);
}
